// SelfAttentionHybrid_65481071410549
// MI455X (gfx1250) — matrix operand order repaired and checked by reading, not compiled, not run on hardware
//
#include <hip/hip_runtime.h>

// ---------------------------------------------------------------------------
// RBF-kernel attention for MI455X (gfx1250, wave32, WMMA bf16 matrix cores)
// B=4, S=2048, D=1024.  Heavy math through v_wmma_f32_16x16x32_bf16, with
// double-buffered async global->LDS staging (ASYNCcnt) when available.
// ---------------------------------------------------------------------------

typedef __attribute__((ext_vector_type(16))) __bf16 v16bf;
typedef __attribute__((ext_vector_type(8)))  float  v8f;
typedef int v4i_vec __attribute__((vector_size(16)));   // matches builtin proto

#define GAMMA 1.0f

#if defined(__gfx1250__) && __has_builtin(__builtin_amdgcn_global_load_async_to_lds_b128)
#define USE_ASYNC_LDS 1
#else
#define USE_ASYNC_LDS 0
#endif

// ---- manual bf16 <-> f32 (storage type: unsigned short) --------------------
__device__ __forceinline__ unsigned short f2bfu(float f) {
    unsigned u = __float_as_uint(f);
    u = (u + 0x7FFFu + ((u >> 16) & 1u)) >> 16;   // round-to-nearest-even
    return (unsigned short)u;
}
__device__ __forceinline__ float bf2f(unsigned short s) {
    return __uint_as_float(((unsigned)s) << 16);
}

// ---- 16B global -> LDS copy: async engine if present, else through VGPRs ---
__device__ __forceinline__ void copy16_g2l(const unsigned short* gsrc,
                                           unsigned short* ldst) {
#if USE_ASYNC_LDS
    __builtin_amdgcn_global_load_async_to_lds_b128(
        (__attribute__((address_space(1))) v4i_vec*)(v4i_vec*)(void*)const_cast<unsigned short*>(gsrc),
        (__attribute__((address_space(3))) v4i_vec*)(v4i_vec*)(void*)ldst,
        0, 0);
#else
    *(uint4*)ldst = *(const uint4*)gsrc;
#endif
}

template <int N>
__device__ __forceinline__ void wait_async() {
#if USE_ASYNC_LDS
#if __has_builtin(__builtin_amdgcn_s_wait_asynccnt)
    __builtin_amdgcn_s_wait_asynccnt(N);
#else
    asm volatile("s_wait_asynccnt %0" ::"i"(N) : "memory");
#endif
#endif
}

// ---- elementwise f32 -> bf16 convert (4 per thread, float4 loads) ----------
__global__ __launch_bounds__(256) void cvt_f32_bf16(const float* __restrict__ src,
                                                    unsigned short* __restrict__ dst,
                                                    long n4) {
    long i = (long)blockIdx.x * blockDim.x + threadIdx.x;
    if (i >= n4) return;
    long b = i * 4;
    float4 f = *(const float4*)(src + b);
    dst[b + 0] = f2bfu(f.x);
    dst[b + 1] = f2bfu(f.y);
    dst[b + 2] = f2bfu(f.z);
    dst[b + 3] = f2bfu(f.w);
}

// ---- W[k][n] (f32) -> Wt[n][k] (bf16) --------------------------------------
__global__ __launch_bounds__(256) void transpose_w(const float* __restrict__ W,
                                                   unsigned short* __restrict__ Wt,
                                                   int D) {
    long idx = (long)blockIdx.x * blockDim.x + threadIdx.x;
    if (idx >= (long)D * D) return;
    int k = (int)(idx / D);
    int n = (int)(idx % D);
    Wt[(long)n * D + k] = f2bfu(W[idx]);
}

// ---- per-row reduce over bf16 matrix: sum or sum-of-squares ----------------
__global__ __launch_bounds__(256) void row_reduce(const unsigned short* __restrict__ X,
                                                  float* __restrict__ out,
                                                  int ncols, int do_sq) {
    int row  = blockIdx.x * 8 + (threadIdx.x >> 5);
    int lane = threadIdx.x & 31;
    const unsigned short* p = X + (long)row * ncols;
    float s = 0.f;
    for (int c = lane; c < ncols; c += 32) {
        float f = bf2f(p[c]);
        s += do_sq ? f * f : f;
    }
    for (int off = 16; off; off >>= 1) s += __shfl_xor(s, off, 32);
    if (lane == 0) out[row] = s;
}

// ---------------------------------------------------------------------------
// Tiled bf16 GEMM core: C[M x N] = A[M x K] * Bt[N x K]^T  (acc f32)
// Block: 256 thr (8 waves), tile 64x128, K-step 32, wave patch 32x32 (2x2 WMMA).
// Double-buffered LDS; staging via async global->LDS (3 x b128 per thread).
// MODE 0: out bf16 = acc + bias[n]                (q/k projection)
// MODE 1: vt[b][n][s] bf16 = acc + bias[n]        (v projection, transposed)
// MODE 2: out bf16 = exp(exp(-g*(qsq+ksq-2acc)))  (unnormalized probs)
// MODE 3: out bf16 = acc / rsum[m]                (attn * V, normalized)
// MODE 4: out f32  = acc + bias[n]                (final projection)
// ---------------------------------------------------------------------------
#define TM 64
#define TN 128
#define TK 32

template <int MODE>
__global__ __launch_bounds__(256) void gemm_bf16(
    const unsigned short* __restrict__ A, int lda,
    const unsigned short* __restrict__ Bt, long b_batch_stride, int ldb,
    int Kdim, int N, int rows_per_batch,
    const float* __restrict__ aux0,     // bias (0,1,4) | qsq (2) | rsum (3)
    const float* __restrict__ aux1,     // ksq (2)
    unsigned short* __restrict__ out_b16,
    float* __restrict__ out_f32,
    int S_out)                          // MODE 1: sequence length
{
    __shared__ __align__(16) unsigned short lds_a[2][TM][TK + 8];
    __shared__ __align__(16) unsigned short lds_b[2][TN][TK + 8];

    const int tid  = threadIdx.x;
    const int m0   = blockIdx.y * TM;
    const int n0   = blockIdx.x * TN;
    const int batch = m0 / rows_per_batch;
    const unsigned short* Bp = Bt + (long)batch * b_batch_stride;

    const int wave = tid >> 5;
    const int lane = tid & 31;
    const int wm = wave >> 2;   // 0..1
    const int wn = wave & 3;    // 0..3

    v8f acc[2][2] = {};

    const int a_r = tid >> 2, a_c = (tid & 3) * 8;    // A stage: 16B / thread
    const int b_r = tid >> 1, b_c = (tid & 1) * 16;   // B stage: 32B / thread
    const int arow = lane & 15;
    const int kb8  = (lane >> 4) * 8;                 // A lane K-base
    const int kb16 = (lane >> 4) * 16;                // unused: B reads at kb8, same k order as A

    const unsigned short* agp = A  + (long)(m0 + a_r) * lda + a_c;
    const unsigned short* bgp = Bp + (long)(n0 + b_r) * ldb + b_c;

    auto stage = [&](int buf, int kk) {
        copy16_g2l(agp + kk, &lds_a[buf][a_r][a_c]);
        copy16_g2l(bgp + kk, &lds_b[buf][b_r][b_c]);
        copy16_g2l(bgp + kk + 8, &lds_b[buf][b_r][b_c + 8]);
    };

    const int nk = Kdim / TK;
    stage(0, 0);

    for (int it = 0; it < nk; ++it) {
        const int cur = it & 1;
        __syncthreads();                       // all readers of buf[1-cur] done
        const bool more = (it + 1 < nk);
        if (more) stage(1 - cur, (it + 1) * TK);
        if (more) wait_async<3>(); else wait_async<0>();
        __syncthreads();                       // buf[cur] visible to all waves

        union Frag { uint4 u[2]; v16bf v; };
        Frag a[2], b[2];
#pragma unroll
        for (int i = 0; i < 2; ++i) {
            int r = wm * 32 + i * 16 + arow;
            a[i].u[0] = *(const uint4*)&lds_a[cur][r][kb8];
            a[i].u[1] = *(const uint4*)&lds_a[cur][r][kb8 + 16];
        }
#pragma unroll
        for (int j = 0; j < 2; ++j) {
            int c = wn * 32 + j * 16 + (lane & 15);
            b[j].u[0] = *(const uint4*)&lds_b[cur][c][kb8];
            b[j].u[1] = *(const uint4*)&lds_b[cur][c][kb8 + 16];
        }
#pragma unroll
        for (int i = 0; i < 2; ++i)
#pragma unroll
            for (int j = 0; j < 2; ++j)
                acc[i][j] = __builtin_amdgcn_wmma_f32_16x16x32_bf16(
                    false, a[i].v, false, b[j].v, (short)0, acc[i][j],
                    false, false);
    }

    // epilogue: C/D layout — VGPR p, lanes 0-15 -> row p, lanes 16-31 -> row p+8
#pragma unroll
    for (int i = 0; i < 2; ++i) {
#pragma unroll
        for (int j = 0; j < 2; ++j) {
            const int mr = m0 + wm * 32 + i * 16 + ((lane >> 4) * 8);
            const int nc = n0 + wn * 32 + j * 16 + (lane & 15);
#pragma unroll
            for (int p = 0; p < 8; ++p) {
                const int m = mr + p;
                const int n = nc;
                float v = acc[i][j][p];
                if (MODE == 0) {
                    v += aux0[n];
                    out_b16[(long)m * N + n] = f2bfu(v);
                } else if (MODE == 1) {
                    v += aux0[n];
                    const int ob = m / S_out;
                    const int s  = m - ob * S_out;
                    out_b16[(long)ob * N * S_out + (long)n * S_out + s] = f2bfu(v);
                } else if (MODE == 2) {
                    const float d2 = aux0[m] + aux1[(long)batch * N + n] - 2.0f * v;
                    const float w  = __expf(__expf(-GAMMA * d2));
                    out_b16[(long)m * N + n] = f2bfu(w);
                } else if (MODE == 3) {
                    out_b16[(long)m * N + n] = f2bfu(v / aux0[m]);
                } else { // MODE 4
                    out_f32[(long)m * N + n] = v + aux0[n];
                }
            }
        }
    }
}

// ---------------------------------------------------------------------------
extern "C" void kernel_launch(void* const* d_in, const int* in_sizes, int n_in,
                              void* d_out, int out_size, void* d_ws, size_t ws_size,
                              hipStream_t stream) {
    const float* x  = (const float*)d_in[0];
    const float* Wq = (const float*)d_in[1];
    const float* bq = (const float*)d_in[2];
    const float* Wk = (const float*)d_in[3];
    const float* bk = (const float*)d_in[4];
    const float* Wv = (const float*)d_in[5];
    const float* bv = (const float*)d_in[6];
    const float* Wo = (const float*)d_in[7];
    const float* bo = (const float*)d_in[8];
    float* out = (float*)d_out;

    const int Bn = 4, S = 2048, D = 1024;
    const int MT = Bn * S;                       // 8192 rows total

    char* base = (char*)d_ws;
    size_t off = 0;
    auto take = [&](size_t bytes) -> void* {
        void* r = base + off;
        off += (bytes + 255) & ~(size_t)255;
        return r;
    };
    unsigned short* xb  = (unsigned short*)take((size_t)MT * D * 2);  // reused as ao
    unsigned short* wqt = (unsigned short*)take((size_t)D * D * 2);
    unsigned short* wkt = (unsigned short*)take((size_t)D * D * 2);
    unsigned short* wvt = (unsigned short*)take((size_t)D * D * 2);
    unsigned short* wot = (unsigned short*)take((size_t)D * D * 2);
    unsigned short* qb  = (unsigned short*)take((size_t)MT * D * 2);
    unsigned short* kbf = (unsigned short*)take((size_t)MT * D * 2);
    unsigned short* vt  = (unsigned short*)take((size_t)Bn * D * S * 2);
    unsigned short* pb  = (unsigned short*)take((size_t)MT * S * 2);
    float* qsq  = (float*)take((size_t)MT * 4);
    float* ksq  = (float*)take((size_t)MT * 4);
    float* rsum = (float*)take((size_t)MT * 4);
    unsigned short* ao = xb;   // x is dead after QKV projections

    // 1) convert x to bf16
    {
        long n4 = (long)MT * D / 4;
        cvt_f32_bf16<<<dim3((unsigned)((n4 + 255) / 256)), 256, 0, stream>>>(x, xb, n4);
    }
    // 2) transpose-convert weights
    {
        dim3 g((unsigned)(((long)D * D + 255) / 256));
        transpose_w<<<g, 256, 0, stream>>>(Wq, wqt, D);
        transpose_w<<<g, 256, 0, stream>>>(Wk, wkt, D);
        transpose_w<<<g, 256, 0, stream>>>(Wv, wvt, D);
        transpose_w<<<g, 256, 0, stream>>>(Wo, wot, D);
    }
    // 3) projections
    {
        dim3 g(D / TN, MT / TM);
        gemm_bf16<0><<<g, 256, 0, stream>>>(xb, D, wqt, 0, D, D, D, MT,
                                            bq, nullptr, qb, nullptr, S);
        gemm_bf16<0><<<g, 256, 0, stream>>>(xb, D, wkt, 0, D, D, D, MT,
                                            bk, nullptr, kbf, nullptr, S);
        gemm_bf16<1><<<g, 256, 0, stream>>>(xb, D, wvt, 0, D, D, D, MT,
                                            bv, nullptr, vt, nullptr, S);
    }
    // 4) row norms
    row_reduce<<<dim3(MT / 8), 256, 0, stream>>>(qb, qsq, D, 1);
    row_reduce<<<dim3(MT / 8), 256, 0, stream>>>(kbf, ksq, D, 1);
    // 5) unnormalized attention weights  w = exp(exp(-g*dist^2)), per batch
    {
        dim3 g(S / TN, MT / TM);
        gemm_bf16<2><<<g, 256, 0, stream>>>(qb, D, kbf, (long)S * D, D, D, S, S,
                                            qsq, ksq, pb, nullptr, S);
    }
    // 6) softmax denominators
    row_reduce<<<dim3(MT / 8), 256, 0, stream>>>(pb, rsum, S, 0);
    // 7) attn @ V with row normalization
    {
        dim3 g(D / TN, MT / TM);
        gemm_bf16<3><<<g, 256, 0, stream>>>(pb, S, vt, (long)D * S, S, S, D, S,
                                            rsum, nullptr, ao, nullptr, S);
    }
    // 8) final projection -> f32 output
    {
        dim3 g(D / TN, MT / TM);
        gemm_bf16<4><<<g, 256, 0, stream>>>(ao, D, wot, 0, D, D, D, MT,
                                            bo, nullptr, nullptr, out, S);
    }
}
